// MRGCO_43069932045114
// MI455X (gfx1250) — matrix operand order repaired and checked by reading, not compiled, not run on hardware
//
#include <hip/hip_runtime.h>

// MRGCO on MI455X (gfx1250): the fft/ifft-of-8 reference collapses to 5
// independent f16-in/f32-acc GEMM chains. HBM floor = 128MB of A read once
// = ~5.5us @ 23.3TB/s; f16 WMMA (16x16x32) keeps compute at the same level.

typedef _Float16 v16h __attribute__((ext_vector_type(16)));
typedef _Float16 v8h  __attribute__((ext_vector_type(8)));
typedef float    v8f  __attribute__((ext_vector_type(8)));
typedef float    v4f  __attribute__((ext_vector_type(4)));

#define N_NODES 2048
#define HID     256

// f[k] = sum_j a[j] * cos(pi*k*j/4), k = 0..4 (rows 5..7 duplicate 3..1)
__device__ __forceinline__ void cos8_fwd(const float a[8], float f[5]) {
  const float s04 = a[0] + a[4], d04 = a[0] - a[4];
  const float s26 = a[2] + a[6];
  const float s15 = a[1] + a[5], d15 = a[1] - a[5];
  const float s37 = a[3] + a[7], d37 = a[3] - a[7];
  const float p = s04 + s26, q = s15 + s37;
  const float t = 0.70710678118654752f * (d15 - d37);
  f[0] = p + q;
  f[1] = d04 + t;
  f[2] = s04 - s26;
  f[3] = d04 - t;
  f[4] = p - q;
}

// Transform (rows x 256 x 8) f32 -> 5 slices of f16 in B-fragment-swizzled
// layout: slice k, m-block mb (32 rows), h-block hb (16 cols):
//   elem(lane, e) = M[mb*32 + 16*(lane/16) + e][hb*16 + lane%16]
// so each lane's 16 f16 (32B) of a WMMA B fragment are contiguous.
__global__ void swizzle_bf16(const float* __restrict__ src,
                             _Float16* __restrict__ dst, int nrows) {
  const int id = blockIdx.x * blockDim.x + threadIdx.x;  // nrows*256 threads
  const int m = id >> 8, h = id & 255;
  const v4f lo = __builtin_nontemporal_load((const v4f*)(src + (size_t)id * 8));
  const v4f hi =
      __builtin_nontemporal_load((const v4f*)(src + (size_t)id * 8) + 1);
  const float a[8] = {lo[0], lo[1], lo[2], lo[3], hi[0], hi[1], hi[2], hi[3]};
  float f[5];
  cos8_fwd(a, f);
  const int mb = m >> 5, rm = m & 31;
  const int g = rm >> 4, e = rm & 15;
  const int lane = (g << 4) | (h & 15);
  const int hb = h >> 4;
  const size_t idx = (((size_t)(mb * 16 + hb)) * 32 + lane) * 16 + e;
  const size_t kstride = (size_t)nrows * 256;
#pragma unroll
  for (int k = 0; k < 5; ++k) dst[(size_t)k * kstride + idx] = (_Float16)f[k];
}

// GEMM1: C1[k] = (A . cosrow_k) @ Xf[k]   (M=2048, K=2048, N=256, k=0..4)
// One 16-wave WG per 16-row M block covers all 256 columns -> A read once
// (non-temporal stream). A->Af conversion done cooperatively into a
// double-buffered LDS tile (2 x 5 x 16x32 f16 = 10KB, one barrier/step).
__global__ void __launch_bounds__(512, 1)
gemm1(const float* __restrict__ A, const _Float16* __restrict__ Xf,
      _Float16* __restrict__ C1) {
  __shared__ __align__(16) _Float16 af[2][5 * 512];
  const int tid  = threadIdx.x;
  const int w    = tid >> 5;    // wave id == conversion row == h-tile
  const int lane = tid & 31;    // == conversion m_local
  const int n0   = blockIdx.x << 4;
  const int rr   = lane & 15, g = lane >> 4;

  const float* aptr = A + (((size_t)(n0 + w) * N_NODES) + lane) * 8;

  v8f acc[5] = {};

  // Prologue: chunk 0 -> buf 0.
  v4f alo = __builtin_nontemporal_load((const v4f*)aptr);
  v4f ahi = __builtin_nontemporal_load((const v4f*)aptr + 1);
  {
    const float a[8] = {alo[0], alo[1], alo[2], alo[3],
                        ahi[0], ahi[1], ahi[2], ahi[3]};
    float f[5];
    cos8_fwd(a, f);
#pragma unroll
    for (int k = 0; k < 5; ++k)
      af[0][k * 512 + w * 32 + lane] = (_Float16)f[k];
  }

  for (int s = 0; s < 64; ++s) {  // K step = 32 over m
    const int cur = s & 1;
    if (s < 63) {  // issue next A chunk early (uniform branch)
      const float* np = aptr + (size_t)(s + 1) * 256;
      alo = __builtin_nontemporal_load((const v4f*)np);
      ahi = __builtin_nontemporal_load((const v4f*)np + 1);
    }
    __syncthreads();  // buf[cur] stores visible; buf[cur^1] reads (s-1) done
#pragma unroll
    for (int k = 0; k < 5; ++k) {
      // A fragment (16x32 f16): lane holds row rr, K chunk [16g..16g+15]
      // -> two 16B LDS loads.
      const v8h flo = *(const v8h*)&af[cur][k * 512 + rr * 32 + g * 16];
      const v8h fhi = *(const v8h*)&af[cur][k * 512 + rr * 32 + g * 16 + 8];
      const v16h afr = __builtin_shufflevector(
          flo, fhi, 0, 1, 2, 3, 4, 5, 6, 7, 8, 9, 10, 11, 12, 13, 14, 15);
      // B fragment: contiguous 32B per lane from swizzled Xf (L2-resident).
      const v16h bfr = *(const v16h*)(
          Xf + ((((size_t)k * 64 + s) * 16 + w) * 32 + lane) * 16);
      acc[k] = __builtin_amdgcn_wmma_f32_16x16x32_f16(
          false, afr, false, bfr, (short)0, acc[k], false, false);
    }
    if (s < 63) {  // convert next chunk into the other buffer
      const float a[8] = {alo[0], alo[1], alo[2], alo[3],
                          ahi[0], ahi[1], ahi[2], ahi[3]};
      float f[5];
      cos8_fwd(a, f);
#pragma unroll
      for (int k = 0; k < 5; ++k)
        af[cur ^ 1][k * 512 + w * 32 + lane] = (_Float16)f[k];
    }
  }

  // Middle ifft->fft pair is identity on symmetric slices: store C1 as f16.
  const int h = (w << 4) + rr;
#pragma unroll
  for (int k = 0; k < 5; ++k) {
    _Float16* cp = C1 + (size_t)k * N_NODES * HID;
#pragma unroll
    for (int j = 0; j < 8; ++j) {
      const int n = n0 + j + (g << 3);  // C/D layout: VGPR j -> M=j+8*(lane/16)
      cp[(size_t)n * HID + h] = (_Float16)acc[k][j];
    }
  }
}

// GEMM2: C2[k] = C1[k] @ Wf[k] (M=2048, K=256, N=256), fused final icos:
//   out[n,h,j] = (1/8) * sum_k C2_k cos(pi*j*k/4), symmetric in j.
__global__ void __launch_bounds__(512, 1)
gemm2(const _Float16* __restrict__ C1, const _Float16* __restrict__ Wf,
      float* __restrict__ out) {
  const int tid  = threadIdx.x;
  const int w    = tid >> 5, lane = tid & 31;
  const int rr   = lane & 15, g = lane >> 4;
  const int n0   = blockIdx.x << 4;

  v8f acc[5] = {};
  for (int s = 0; s < 8; ++s) {  // K step = 32 over h'
#pragma unroll
    for (int k = 0; k < 5; ++k) {
      const _Float16* ap =
          C1 + (size_t)k * N_NODES * HID + (size_t)(n0 + rr) * HID + s * 32;
      const v8h flo = *(const v8h*)(ap + g * 16);
      const v8h fhi = *(const v8h*)(ap + g * 16 + 8);
      const v16h afr = __builtin_shufflevector(
          flo, fhi, 0, 1, 2, 3, 4, 5, 6, 7, 8, 9, 10, 11, 12, 13, 14, 15);
      const v16h bfr = *(const v16h*)(
          Wf + ((((size_t)k * 8 + s) * 16 + w) * 32 + lane) * 16);
      acc[k] = __builtin_amdgcn_wmma_f32_16x16x32_f16(
          false, afr, false, bfr, (short)0, acc[k], false, false);
    }
  }

  const int h = (w << 4) + rr;
#pragma unroll
  for (int j = 0; j < 8; ++j) {
    const int n = n0 + j + (g << 3);
    const float c0 = acc[0][j], c1 = acc[1][j], c2 = acc[2][j],
                c3 = acc[3][j], c4 = acc[4][j];
    const float e04 = c0 + c4, o04 = c0 - c4;
    const float t = 1.41421356237309505f * (c1 - c3);
    const float y0 = 0.125f * (e04 + 2.0f * (c1 + c2 + c3));
    const float y1 = 0.125f * (o04 + t);
    const float y2 = 0.125f * (e04 - 2.0f * c2);
    const float y3 = 0.125f * (o04 - t);
    const float y4 = 0.125f * (e04 - 2.0f * (c1 - c2 + c3));
    float* op = out + ((size_t)n * HID + h) * 8;
    const v4f r0 = {y0, y1, y2, y3};
    const v4f r1 = {y4, y3, y2, y1};  // relation axis is symmetric
    *(v4f*)op       = r0;
    *((v4f*)op + 1) = r1;
  }
}

extern "C" void kernel_launch(void* const* d_in, const int* in_sizes, int n_in,
                              void* d_out, int out_size, void* d_ws,
                              size_t ws_size, hipStream_t stream) {
  const float* X = (const float*)d_in[0];  // (2048, 256, 8)
  const float* A = (const float*)d_in[1];  // (2048, 2048, 8)
  const float* W = (const float*)d_in[2];  // (256, 256, 8)
  float* out = (float*)d_out;              // (2048, 256, 8)

  _Float16* Xf = (_Float16*)d_ws;                 // 5 * 2048*256 f16 (5MB)
  _Float16* Wf = Xf + (size_t)5 * N_NODES * HID;  // 5 * 256*256  f16 (640KB)
  _Float16* C1 = Wf + (size_t)5 * HID * HID;      // 5 * 2048*256 f16 (5MB)

  hipLaunchKernelGGL(swizzle_bf16, dim3((N_NODES * HID) / 256), dim3(256), 0,
                     stream, X, Xf, N_NODES);
  hipLaunchKernelGGL(swizzle_bf16, dim3((HID * HID) / 256), dim3(256), 0,
                     stream, W, Wf, HID);
  hipLaunchKernelGGL(gemm1, dim3(N_NODES / 16), dim3(512), 0, stream, A, Xf,
                     C1);
  hipLaunchKernelGGL(gemm2, dim3(N_NODES / 16), dim3(512), 0, stream, C1, Wf,
                     out);
}
